// QInferenceWrapper_2241972928514
// MI455X (gfx1250) — matrix operand order repaired and checked by reading, not compiled, not run on hardware
//
#include <hip/hip_runtime.h>
#include <math.h>

typedef __attribute__((ext_vector_type(16))) _Float16 v16h;
typedef __attribute__((ext_vector_type(8)))  float    v8f;

#define T_STEPS 2048
#define HD      512     // H
#define G4H     2048    // 4*H
#define IN2     1024    // 2*I
#define NWG     16      // workgroups per direction in recurrence
#define REC_BLOCK 256
#define WROWS   128     // W_hh rows per WG (4 gates * 32 h-columns)

__device__ __forceinline__ float sigmoidf(float v) {
    return 1.0f / (1.0f + expf(-v));
}

// ---------------- init: zero h buffer and step counters (graph-replay safe) ----
__global__ void init_kernel(float* hbuf, int* ctr) {
    for (int i = threadIdx.x; i < 2 * HD; i += blockDim.x) hbuf[i] = 0.0f;
    if (threadIdx.x < 2) ctr[threadIdx.x] = 0;
}

// ---------------- phase 1: pre = inp0 @ W_ih^T + (b_ih + b_hh)  (WMMA f16) ----
// inp0[t, i] = (i < 512) ? x[t,0,i] : y[t,0,i-512];  x,y are (T, 8, 512)
// W is (4H, 2I) row-major; pre is (T, 4H) row-major.
__global__ void pre_gemm_kernel(const float* __restrict__ x,
                                const float* __restrict__ y,
                                const float* __restrict__ W,
                                const float* __restrict__ bi,
                                const float* __restrict__ bh,
                                float* __restrict__ pre) {
    const int lane  = threadIdx.x & 31;
    const int wid   = threadIdx.x >> 5;
    const int n0    = (blockIdx.x * 8 + wid) * 16;   // output column tile
    const int m0    = blockIdx.y * 16;               // output row tile (time)

    const int lhalf = (lane < 16) ? 0 : 1;
    const int lidx  = lane & 15;
    const int arow  = m0 + lidx;                     // A row this lane carries
    const int bcol  = n0 + lidx;                     // B column this lane carries
    const int aoffK = lhalf ? 16 : 0;                // contiguous K order for A: k = 16h + s
    const int boffK = lhalf ? 16 : 0;                // contiguous K order for B: k = 16h + s

    v8f acc = {};

    #pragma unroll 2
    for (int k0 = 0; k0 < IN2; k0 += 32) {
        const float* src = (k0 < HD) ? x : y;        // 32-chunk never crosses x/y seam
        const int    kb  = (k0 < HD) ? k0 : (k0 - HD);
        const float* ap  = src + (size_t)arow * 4096 + kb;   // batch row 0, stride B*I
        const float* bp  = W + (size_t)bcol * IN2 + k0 + boffK;

        v16h a, b;
        #pragma unroll
        for (int e = 0; e < 16; ++e) {
            int ka = aoffK + e;                      // lanes<16: K 0..15 ; lanes>=16: K 16..31
            a[e] = (_Float16)ap[ka];
        }
        #pragma unroll
        for (int e = 0; e < 16; ++e) {
            b[e] = (_Float16)bp[e];                  // lanes<16: K 0..15 ; lanes>=16: K 16..31
        }
        acc = __builtin_amdgcn_wmma_f32_16x16x32_f16(false, a, false, b,
                                                     (short)0, acc, false, false);
    }

    const int   col  = n0 + lidx;
    const float bias = bi[col] + bh[col];
    #pragma unroll
    for (int r = 0; r < 8; ++r) {
        int row = m0 + r + (lhalf ? 8 : 0);          // C/D layout: VGPR r -> M=r / M=8+r
        pre[(size_t)row * G4H + col] = acc[r] + bias;
    }
}

// ---------------- phase 2: serial LSTM scan, both directions ------------------
// 32 persistent WGs: dir = bid>>4, slice k = bid&15 owns h columns [32k,32k+32).
// Each WG holds its 128 W_hh rows (fp32, 256 KB) in LDS; h exchanged via L2
// with an agent-scope atomic step barrier.
__global__ void lstm_recurrence_kernel(const float* __restrict__ pre_f,
                                       const float* __restrict__ pre_r,
                                       const float* __restrict__ Whh_f,
                                       const float* __restrict__ Whh_r,
                                       float* __restrict__ hbuf,   // 2*512
                                       int*   __restrict__ ctr,    // 2
                                       float* __restrict__ ctx) {  // (T, 2H)
    extern __shared__ float lds[];
    float* Wl = lds;                 // WROWS * HD
    float* hl = Wl + WROWS * HD;     // HD
    float* gl = hl + HD;             // WROWS

    const int tid = threadIdx.x;
    const int bid = blockIdx.x;
    const int dir = bid >> 4;
    const int k   = bid & 15;

    const float* W   = dir ? Whh_r : Whh_f;
    const float* pre = dir ? pre_r : pre_f;
    float* hb = hbuf + dir * HD;
    int*   cd = ctr + dir;

    // stage this WG's W_hh rows: local row rr -> global row (rr/32)*512 + 32k + rr%32
    for (int idx = tid; idx < WROWS * HD; idx += REC_BLOCK) {
        int rr  = idx >> 9;
        int col = idx & 511;
        int g   = rr >> 5;
        int j   = rr & 31;
        Wl[idx] = W[(size_t)(g * HD + k * 32 + j) * HD + col];
    }
    for (int i = tid; i < HD; i += REC_BLOCK) hl[i] = 0.0f;  // h0 = 0
    __syncthreads();

    float cst = 0.0f;                 // cell state for column (32k + tid), tid < 32

    const int rr   = tid >> 1;        // two lanes per dot-product row
    const int half = tid & 1;
    const float* wrow = Wl + rr * HD + half * 256;
    const int grow = (rr >> 5) * HD + k * 32 + (rr & 31);

    for (int s = 0; s < T_STEPS; ++s) {
        const int t = dir ? (T_STEPS - 1 - s) : s;

        // 128 dots of length 512, split in halves across lane pairs
        const float* hp = hl + half * 256;
        float acc = 0.0f;
        #pragma unroll 8
        for (int i = 0; i < 256; ++i) acc += wrow[i] * hp[i];
        acc += __shfl_xor(acc, 1, 32);
        if (half == 0)
            gl[rr] = acc + pre[(size_t)t * G4H + grow];
        __syncthreads();

        if (tid < 32) {               // gate order i, f, g, o
            float gi = sigmoidf(gl[tid]);
            float gf = sigmoidf(gl[32 + tid]);
            float gg = tanhf(gl[64 + tid]);
            float go = sigmoidf(gl[96 + tid]);
            cst = gf * cst + gi * gg;
            float h = go * tanhf(cst);
            int col = k * 32 + tid;
            hb[col] = h;
            ctx[(size_t)t * (2 * HD) + dir * HD + col] = h;
        }
        __threadfence();
        __syncthreads();

        if (tid == 0) {               // step barrier across this direction's 16 WGs
            __hip_atomic_fetch_add(cd, 1, __ATOMIC_RELEASE, __HIP_MEMORY_SCOPE_AGENT);
            const int target = NWG * (s + 1);
            while (__hip_atomic_load(cd, __ATOMIC_ACQUIRE, __HIP_MEMORY_SCOPE_AGENT) < target)
                __builtin_amdgcn_s_sleep(1);
        }
        __syncthreads();

        // pull the full updated h (all 16 slices) with agent-scope loads
        for (int i = tid; i < HD; i += REC_BLOCK)
            hl[i] = __hip_atomic_load(&hb[i], __ATOMIC_RELAXED, __HIP_MEMORY_SCOPE_AGENT);
        __syncthreads();
    }
}

// ---------------- phase 3: z = sigmoid(relu(ctx @ fc_w^T + fc_b)) -------------
__global__ void fc_kernel(const float* __restrict__ ctx,
                          const float* __restrict__ fcw,  // (H, 2H)
                          const float* __restrict__ fcb,
                          float* __restrict__ out) {      // (T, 1, H)
    const int m = blockIdx.x * blockDim.x + threadIdx.x;
    const int t = blockIdx.y;
    if (m >= HD) return;
    const float* cr = ctx + (size_t)t * (2 * HD);
    const float* wr = fcw + (size_t)m * (2 * HD);
    float s = fcb[m];
    #pragma unroll 4
    for (int j = 0; j < 2 * HD; ++j) s += cr[j] * wr[j];
    float r = (s > 0.0f) ? s : 0.0f;
    out[(size_t)t * HD + m] = 1.0f / (1.0f + expf(-r));
}

extern "C" void kernel_launch(void* const* d_in, const int* in_sizes, int n_in,
                              void* d_out, int out_size, void* d_ws, size_t ws_size,
                              hipStream_t stream) {
    (void)in_sizes; (void)n_in; (void)out_size; (void)ws_size;

    const float* x      = (const float*)d_in[0];
    const float* y      = (const float*)d_in[1];
    const float* W_ih_f = (const float*)d_in[2];
    const float* W_hh_f = (const float*)d_in[3];
    const float* b_ih_f = (const float*)d_in[4];
    const float* b_hh_f = (const float*)d_in[5];
    const float* W_ih_r = (const float*)d_in[6];
    const float* W_hh_r = (const float*)d_in[7];
    const float* b_ih_r = (const float*)d_in[8];
    const float* b_hh_r = (const float*)d_in[9];
    const float* fc_w   = (const float*)d_in[10];
    const float* fc_b   = (const float*)d_in[11];
    float* out = (float*)d_out;

    char* ws = (char*)d_ws;
    float* pre_f = (float*)ws;  ws += sizeof(float) * (size_t)T_STEPS * G4H;
    float* pre_r = (float*)ws;  ws += sizeof(float) * (size_t)T_STEPS * G4H;
    float* ctx   = (float*)ws;  ws += sizeof(float) * (size_t)T_STEPS * 2 * HD;
    float* hbuf  = (float*)ws;  ws += sizeof(float) * 2 * HD;
    int*   ctr   = (int*)ws;

    init_kernel<<<1, 256, 0, stream>>>(hbuf, ctr);

    // pre-activation GEMMs, one per direction: grid (2048/128 cols, 2048/16 rows)
    dim3 gg(16, 128);
    pre_gemm_kernel<<<gg, 256, 0, stream>>>(x, y, W_ih_f, b_ih_f, b_hh_f, pre_f);
    pre_gemm_kernel<<<gg, 256, 0, stream>>>(x, y, W_ih_r, b_ih_r, b_hh_r, pre_r);

    // persistent serial scan: 16 WGs per direction, 256 KB + h + gates in LDS
    size_t lds_bytes = sizeof(float) * (WROWS * HD + HD + WROWS);
    lstm_recurrence_kernel<<<2 * NWG, REC_BLOCK, lds_bytes, stream>>>(
        pre_f, pre_r, W_hh_f, W_hh_r, hbuf, ctr, ctx);

    fc_kernel<<<dim3(HD / 256, T_STEPS), 256, 0, stream>>>(ctx, fc_w, fc_b, out);
}
